// SimpleCloudVoxelizer_69630009803469
// MI455X (gfx1250) — compile-verified, matrix operand order checked by reading, not run on hardware
//
#include <hip/hip_runtime.h>
#include <cstdint>

// ---------------------------------------------------------------------------
// SimpleCloudVoxelizer for MI455X (gfx1250, wave32).
// HBM floor: 2 streaming passes over 96MB points (~9us @ 23.3TB/s).
// All voxel-sized arrays (<= 2.8MB each) are L2-resident (192MB L2).
// FPS (409 serial steps x 352K candidates) uses v_wmma_f32_16x16x32_f16 to
// batch 16 candidate-vs-pivot dot products per wave per issue.
// Point streaming uses double-buffered async global->LDS copies (ASYNCcnt).
// ---------------------------------------------------------------------------

#define MAX_PTS    32
#define MAX_VOXELS 1024
#define P_DENSEST  614
#define Q_FPS      410
#define NVOX_MAX   360000

typedef __attribute__((ext_vector_type(16))) _Float16 v16h;
typedef __attribute__((ext_vector_type(8)))  float    v8f;

// async-copy builtins take int pointers in AS1 (global) / AS3 (LDS)
#define AS1INT(x) ((__attribute__((address_space(1))) int*)(uintptr_t)(x))
#define AS3INT(x) ((__attribute__((address_space(3))) int*)(x))

#if __has_builtin(__builtin_amdgcn_global_load_async_to_lds_b32)
#define HAVE_ASYNC_LDS 1
#else
#define HAVE_ASYNC_LDS 0
#endif

#if __has_builtin(__builtin_amdgcn_s_wait_asynccnt)
#define WAIT_ASYNC(n) __builtin_amdgcn_s_wait_asynccnt(n)
#else
#define WAIT_ASYNC(n) asm volatile("s_wait_asynccnt %0" ::"i"(n) : "memory")
#endif

// ---------------------------------------------------------------------------

__device__ inline void load_params(const float* vs, const float* cr,
                                   float3& vsz, float3& org,
                                   int& gx, int& gy, int& gz) {
  vsz = make_float3(vs[0], vs[1], vs[2]);
  org = make_float3(cr[0], cr[1], cr[2]);
  gx = (int)floorf((cr[3] - cr[0]) / vs[0]);
  gy = (int)floorf((cr[4] - cr[1]) / vs[1]);
  gz = (int)floorf((cr[5] - cr[2]) / vs[2]);
}

__device__ inline int voxel_key(float x, float y, float z,
                                float3 vsz, float3 org,
                                int gx, int gy, int gz) {
  int ix = (int)floorf((x - org.x) / vsz.x);
  int iy = (int)floorf((y - org.y) / vsz.y);
  int iz = (int)floorf((z - org.z) / vsz.z);
  ix = min(max(ix, 0), gx - 1);
  iy = min(max(iy, 0), gy - 1);
  iz = min(max(iz, 0), gz - 1);
  return (ix * gy + iy) * gz + iz;
}

// ---------------------------------------------------------------------------
// K0: zero/initialize workspace + output (output is poisoned by harness).
__global__ void k_init(unsigned* __restrict__ counts, unsigned* __restrict__ cursor,
                       int* __restrict__ new_index, float* __restrict__ dist,
                       float* __restrict__ out, int out_n) {
  const int stride = gridDim.x * blockDim.x;
  for (int i = blockIdx.x * blockDim.x + threadIdx.x; i < NVOX_MAX; i += stride) {
    counts[i] = 0u;
    cursor[i] = 0u;
    new_index[i] = -1;
    dist[i] = 3.0e38f;
  }
  for (int i = blockIdx.x * blockDim.x + threadIdx.x; i < out_n; i += stride) {
    out[i] = 0.0f;
  }
}

// ---------------------------------------------------------------------------
// K1: histogram of points per voxel. Streaming 96MB; double-buffered async
// global->LDS staging hides HBM latency without VGPR pressure.
#define HBLK 256
__global__ __launch_bounds__(HBLK) void k_hist(const float* __restrict__ pts, long long N,
                                               const float* __restrict__ vs,
                                               const float* __restrict__ cr,
                                               unsigned* __restrict__ counts) {
  float3 vsz, org; int gx, gy, gz;
  load_params(vs, cr, vsz, org, gx, gy, gz);
  const int tid = threadIdx.x;

#if HAVE_ASYNC_LDS
  __shared__ float sbuf[2][HBLK * 3];
  const long long nchunks = (N + HBLK - 1) / HBLK;
  long long chunk = blockIdx.x;
  if (chunk < nchunks) {
    { // stage first chunk into buffer 0 (3 async b32 ops per wave issue-slot)
      long long i = chunk * (long long)HBLK + tid;
      if (i < N) {
        const float* g = pts + i * 3;
        __builtin_amdgcn_global_load_async_to_lds_b32(AS1INT(g + 0), AS3INT(&sbuf[0][tid * 3 + 0]), 0, 0);
        __builtin_amdgcn_global_load_async_to_lds_b32(AS1INT(g + 1), AS3INT(&sbuf[0][tid * 3 + 1]), 0, 0);
        __builtin_amdgcn_global_load_async_to_lds_b32(AS1INT(g + 2), AS3INT(&sbuf[0][tid * 3 + 2]), 0, 0);
      }
    }
    int buf = 0;
    for (; chunk < nchunks; chunk += gridDim.x, buf ^= 1) {
      const long long nxt = chunk + gridDim.x;
      const bool have_next = (nxt < nchunks);   // uniform across block
      if (have_next) {
        long long i = nxt * (long long)HBLK + tid;
        if (i < N) {
          const float* g = pts + i * 3;
          __builtin_amdgcn_global_load_async_to_lds_b32(AS1INT(g + 0), AS3INT(&sbuf[buf ^ 1][tid * 3 + 0]), 0, 0);
          __builtin_amdgcn_global_load_async_to_lds_b32(AS1INT(g + 1), AS3INT(&sbuf[buf ^ 1][tid * 3 + 1]), 0, 0);
          __builtin_amdgcn_global_load_async_to_lds_b32(AS1INT(g + 2), AS3INT(&sbuf[buf ^ 1][tid * 3 + 2]), 0, 0);
        }
      }
      // async loads complete in order: waiting <=3 drains the older stage only
      if (have_next) { WAIT_ASYNC(3); } else { WAIT_ASYNC(0); }
      __syncthreads();
      long long i = chunk * (long long)HBLK + tid;
      if (i < N) {
        float x = sbuf[buf][tid * 3 + 0];
        float y = sbuf[buf][tid * 3 + 1];
        float z = sbuf[buf][tid * 3 + 2];
        atomicAdd(&counts[voxel_key(x, y, z, vsz, org, gx, gy, gz)], 1u);
      }
      __syncthreads();  // buffer consumed before it is restaged
    }
  }
#else
  const long long stride = (long long)gridDim.x * blockDim.x;
  for (long long i = (long long)blockIdx.x * blockDim.x + tid; i < N; i += stride) {
    const float* g = pts + i * 3;
    atomicAdd(&counts[voxel_key(g[0], g[1], g[2], vsz, org, gx, gy, gz)], 1u);
  }
#endif
}

// ---------------------------------------------------------------------------
// K2: pack centroid of every voxel as 4 x f16 {x,y,z,0} (WMMA A/B fragment food).
__global__ void k_cent(unsigned long long* __restrict__ cent,
                       const float* __restrict__ vs, const float* __restrict__ cr) {
  float3 vsz, org; int gx, gy, gz;
  load_params(vs, cr, vsz, org, gx, gy, gz);
  const int nvox = gx * gy * gz;
  const int stride = gridDim.x * blockDim.x;
  for (int i = blockIdx.x * blockDim.x + threadIdx.x; i < nvox; i += stride) {
    int iz = i % gz;
    int iy = (i / gz) % gy;
    int ix = i / (gy * gz);
    union { unsigned long long u; _Float16 h[4]; } p;
    p.h[0] = (_Float16)(ix * vsz.x + org.x);
    p.h[1] = (_Float16)(iy * vsz.y + org.y);
    p.h[2] = (_Float16)(iz * vsz.z + org.z);
    p.h[3] = (_Float16)0.0f;
    cent[i] = p.u;
  }
}

// ---------------------------------------------------------------------------
// K3: iterative top-614 densest (single WG; counts array is L2-resident).
__global__ __launch_bounds__(1024) void k_topk(const unsigned* __restrict__ counts,
                                               int* __restrict__ new_index,
                                               int* __restrict__ sel_keys,
                                               const float* __restrict__ vs,
                                               const float* __restrict__ cr) {
  float3 vsz, org; int gx, gy, gz;
  load_params(vs, cr, vsz, org, gx, gy, gz);
  const int nvox = gx * gy * gz;
  const int tid = threadIdx.x;
  __shared__ unsigned s_c[1024];
  __shared__ int s_k[1024];
  for (int s = 0; s < P_DENSEST; ++s) {
    unsigned bc = 0u;
    int bk = 0x7fffffff;
    for (int i = tid; i < nvox; i += 1024) {
      if (new_index[i] != -1) continue;
      unsigned c = counts[i];
      if (c > bc || (c == bc && i < bk)) { bc = c; bk = i; }
    }
    s_c[tid] = bc; s_k[tid] = bk;
    __syncthreads();
    for (int off = 512; off > 0; off >>= 1) {
      if (tid < off) {
        unsigned c2 = s_c[tid + off]; int k2 = s_k[tid + off];
        if (c2 > s_c[tid] || (c2 == s_c[tid] && k2 < s_k[tid])) { s_c[tid] = c2; s_k[tid] = k2; }
      }
      __syncthreads();
    }
    if (tid == 0 && s_k[0] < nvox) {
      new_index[s_k[0]] = s;
      sel_keys[s] = s_k[0];
    }
    __threadfence_block();
    __syncthreads();
  }
}

// ---------------------------------------------------------------------------
// K4: farthest-point sampling of 410 voxels. Single WG of 32 waves; each wave
// runs v_wmma_f32_16x16x32_f16 on tiles of 16 candidate centroids against the
// broadcast pivot. 16-bit A layout: lanes 0-15 hold row M=lane, halves
// [0..1]=(K0,K1)=(x,y), [2..3]=(K2,K3)=(z,0); lanes 16-31 cover K>=8 -> zero.
// The all-columns-equal broadcast B fragment has the identical shape.
// D layout: lanes 0-15 vgpr j = D[M=j][N=lane]; lanes 16-31 vgpr j = D[M=8+j].
__global__ __launch_bounds__(1024) void k_fps(const unsigned* __restrict__ counts,
                                              int* __restrict__ new_index,
                                              int* __restrict__ sel_keys,
                                              float* __restrict__ dist,
                                              const unsigned long long* __restrict__ cent,
                                              const float* __restrict__ vs,
                                              const float* __restrict__ cr) {
  float3 vsz, org; int gx, gy, gz;
  load_params(vs, cr, vsz, org, gx, gy, gz);
  const int nvox = gx * gy * gz;
  const int tid  = threadIdx.x;
  const int lane = tid & 31;
  const int wave = tid >> 5;

  __shared__ float s_last[3];
  __shared__ float s_val[1024];
  __shared__ int   s_idx[1024];

  // --- pick start: first remaining voxel (argmax over boolean mask) ---
  int first = 0x7fffffff;
  for (int i = tid; i < nvox; i += 1024)
    if (new_index[i] == -1 && counts[i] > 0u && i < first) first = i;
  s_idx[tid] = first;
  __syncthreads();
  for (int off = 512; off > 0; off >>= 1) {
    if (tid < off) s_idx[tid] = min(s_idx[tid], s_idx[tid + off]);
    __syncthreads();
  }
  if (tid == 0) {
    int b = min(s_idx[0], nvox - 1);
    new_index[b] = P_DENSEST;
    sel_keys[P_DENSEST] = b;
    union { unsigned long long u; _Float16 h[4]; } p; p.u = cent[b];
    s_last[0] = (float)p.h[0]; s_last[1] = (float)p.h[1]; s_last[2] = (float)p.h[2];
  }
  __threadfence_block();
  __syncthreads();

  const int ntiles = (nvox + 15) / 16;

  for (int q = 1; q < Q_FPS; ++q) {
    const float lx = s_last[0], ly = s_last[1], lz = s_last[2];
    const float ln2 = lx * lx + ly * ly + lz * lz;
    v16h bfrag = {};
    if (lane < 16) {
      bfrag[0] = (_Float16)lx; bfrag[1] = (_Float16)ly; bfrag[2] = (_Float16)lz;
    }
    float bestv = -3.0e38f;
    int   besti = 0x7fffffff;

    for (int t = wave; t < ntiles; t += 32) {
      const int base = t * 16;
      v16h afrag = {};
      if (lane < 16) {
        int idx = base + lane;
        if (idx < nvox) {
          union { unsigned long long u; _Float16 h[4]; } p; p.u = cent[idx];
          afrag[0] = p.h[0]; afrag[1] = p.h[1]; afrag[2] = p.h[2];
        }
      }
      v8f acc = {};
      acc = __builtin_amdgcn_wmma_f32_16x16x32_f16(
          /*neg_a=*/false, afrag, /*neg_b=*/false, bfrag,
          /*c_mod=*/(short)0, acc, /*reuse_a=*/false, /*reuse_b=*/false);

      // owner lanes: 0-7 -> rows 0-7 (vgpr=lane); 16-23 -> rows 8-15 (vgpr=lane-16)
      int r = (lane < 8) ? lane : ((lane >= 16 && lane < 24) ? 8 + (lane - 16) : -1);
      if (r >= 0) {
        int idx = base + r;
        if (idx < nvox) {
          int j = lane & 7;
          float dot = 0.0f;
#pragma unroll
          for (int k = 0; k < 8; ++k)
            if (j == k) dot = acc[k];
          union { unsigned long long u; _Float16 h[4]; } p; p.u = cent[idx];
          float cx = (float)p.h[0], cy = (float)p.h[1], cz = (float)p.h[2];
          float d  = (cx * cx + cy * cy + cz * cz) - 2.0f * dot + ln2;
          float nd = fminf(dist[idx], d);
          dist[idx] = nd;
          bool valid = (new_index[idx] == -1) && (counts[idx] > 0u);
          float v = valid ? nd : -3.0e38f;
          if (v > bestv || (v == bestv && idx < besti)) { bestv = v; besti = idx; }
        }
      }
    }

    s_val[tid] = bestv; s_idx[tid] = besti;
    __syncthreads();
    for (int off = 512; off > 0; off >>= 1) {
      if (tid < off) {
        float v2 = s_val[tid + off]; int i2 = s_idx[tid + off];
        if (v2 > s_val[tid] || (v2 == s_val[tid] && i2 < s_idx[tid])) {
          s_val[tid] = v2; s_idx[tid] = i2;
        }
      }
      __syncthreads();
    }
    if (tid == 0) {
      int b = (s_idx[0] < nvox) ? s_idx[0] : 0;
      new_index[b] = P_DENSEST + q;
      sel_keys[P_DENSEST + q] = b;
      union { unsigned long long u; _Float16 h[4]; } p; p.u = cent[b];
      s_last[0] = (float)p.h[0]; s_last[1] = (float)p.h[1]; s_last[2] = (float)p.h[2];
    }
    __threadfence_block();
    __syncthreads();
  }
}

// ---------------------------------------------------------------------------
// K5: scatter points into selected voxel slots (second 96MB streaming pass).
__global__ void k_scatter(const float* __restrict__ pts, long long N,
                          const float* __restrict__ vs, const float* __restrict__ cr,
                          const int* __restrict__ new_index,
                          unsigned* __restrict__ cursor,
                          float* __restrict__ out) {
  float3 vsz, org; int gx, gy, gz;
  load_params(vs, cr, vsz, org, gx, gy, gz);
  const long long stride = (long long)gridDim.x * blockDim.x;
  for (long long i = (long long)blockIdx.x * blockDim.x + threadIdx.x; i < N; i += stride) {
    const float* g = pts + i * 3;
    __builtin_prefetch(pts + (i + stride) * 3, 0, 1);  // global_prefetch_b8 (speculative)
    float x = g[0], y = g[1], z = g[2];
    int key  = voxel_key(x, y, z, vsz, org, gx, gy, gz);
    int slot = new_index[key];
    if (slot >= 0) {
      unsigned r = atomicAdd(&cursor[key], 1u);
      if (r < MAX_PTS) {
        float* d = out + ((size_t)slot * MAX_PTS + r) * 3;
        d[0] = x; d[1] = y; d[2] = z;
      }
    }
  }
}

// ---------------------------------------------------------------------------
// K6: coords + nums for the 1024 selected voxels.
__global__ void k_final(const unsigned* __restrict__ counts,
                        const int* __restrict__ sel_keys,
                        const float* __restrict__ vs, const float* __restrict__ cr,
                        float* __restrict__ out) {
  float3 vsz, org; int gx, gy, gz;
  load_params(vs, cr, vsz, org, gx, gy, gz);
  int i = blockIdx.x * blockDim.x + threadIdx.x;
  if (i >= MAX_VOXELS) return;
  int key = sel_keys[i];
  key = min(max(key, 0), gx * gy * gz - 1);
  int iz = key % gz;
  int iy = (key / gz) % gy;
  int ix = key / (gy * gz);
  float* cptr = out + (size_t)MAX_VOXELS * MAX_PTS * 3 + (size_t)i * 3;
  cptr[0] = (float)ix; cptr[1] = (float)iy; cptr[2] = (float)iz;
  unsigned c = counts[key];
  out[(size_t)MAX_VOXELS * MAX_PTS * 3 + (size_t)MAX_VOXELS * 3 + i] =
      (float)((c < (unsigned)MAX_PTS) ? c : (unsigned)MAX_PTS);
}

// ---------------------------------------------------------------------------

extern "C" void kernel_launch(void* const* d_in, const int* in_sizes, int n_in,
                              void* d_out, int out_size, void* d_ws, size_t ws_size,
                              hipStream_t stream) {
  (void)n_in; (void)ws_size;
  const float* pts = (const float*)d_in[0];
  const float* vs  = (const float*)d_in[1];
  const float* cr  = (const float*)d_in[2];
  const long long N = (long long)in_sizes[0] / 3;
  float* out = (float*)d_out;

  // Workspace layout (~9.0 MB total):
  char* ws = (char*)d_ws;
  unsigned*           counts    = (unsigned*)ws;                         // 1.44 MB
  unsigned*           cursor    = counts + NVOX_MAX;                     // 1.44 MB
  int*                new_index = (int*)(cursor + NVOX_MAX);             // 1.44 MB
  float*              dist      = (float*)(new_index + NVOX_MAX);        // 1.44 MB
  unsigned long long* cent      = (unsigned long long*)(dist + NVOX_MAX);// 2.88 MB
  int*                sel_keys  = (int*)(cent + NVOX_MAX);               // 4 KB

  k_init<<<1024, 256, 0, stream>>>(counts, cursor, new_index, dist, out, out_size);
  k_hist<<<2048, HBLK, 0, stream>>>(pts, N, vs, cr, counts);
  k_cent<<<1024, 256, 0, stream>>>(cent, vs, cr);
  k_topk<<<1, 1024, 0, stream>>>(counts, new_index, sel_keys, vs, cr);
  k_fps<<<1, 1024, 0, stream>>>(counts, new_index, sel_keys, dist, cent, vs, cr);
  k_scatter<<<4096, 256, 0, stream>>>(pts, N, vs, cr, new_index, cursor, out);
  k_final<<<(MAX_VOXELS + 255) / 256, 256, 0, stream>>>(counts, sel_keys, vs, cr, out);
}
